// DecoderMoEModel_44736379355139
// MI455X (gfx1250) — matrix operand order repaired and checked by reading, not compiled, not run on hardware
//
#include <hip/hip_runtime.h>
#include <hip/hip_bf16.h>
#include <math.h>

// ---------------- model dims ----------------
#define Bb   2
#define Tt   1024
#define Dd   1024
#define Hh   16
#define Ll   2
#define Ee   8
#define HIDd 2048
#define Vv   32000
#define HDd  64
#define NTOK (Bb*Tt)
#define EPSf 1e-6f

// ---------------- WMMA types ----------------
typedef _Float16 h4   __attribute__((ext_vector_type(4)));
typedef _Float16 h8   __attribute__((ext_vector_type(8)));
typedef _Float16 v16h __attribute__((ext_vector_type(16)));
typedef float    v8f  __attribute__((ext_vector_type(8)));

union Frag { v16h v; h8 h[2]; };

__device__ __forceinline__ v8f wmma16(const Frag& a, const Frag& b, v8f c) {
  // D = A(16x32 f16) * B(32x16 f16) + C(16x16 f32)
  return __builtin_amdgcn_wmma_f32_16x16x32_f16(false, a.v, false, b.v,
                                                (short)0, c, false, false);
}

// ---- CDNA5 async global->LDS copy (16B per lane), tracked by ASYNCcnt ----
__device__ __forceinline__ void async_copy_b128(unsigned lds_off, const void* gaddr) {
  asm volatile("global_load_async_to_lds_b128 %0, %1, off"
               :: "v"(lds_off), "v"(gaddr) : "memory");
}
__device__ __forceinline__ void wait_async0() {
#if defined(__has_builtin) && __has_builtin(__builtin_amdgcn_s_wait_asynccnt)
  __builtin_amdgcn_s_wait_asynccnt(0);
#else
  asm volatile("s_wait_asynccnt 0" ::: "memory");
#endif
}
__device__ __forceinline__ unsigned lds_off_of(const void* p) {
  return (unsigned)(uintptr_t)p;   // low 32 bits of flat LDS addr = LDS offset
}

// =====================================================================
// Embedding gather: x[n,d] = tok_embed[ids[n], d]   (f32)
// =====================================================================
__global__ __launch_bounds__(256) void embed_kernel(const int* __restrict__ ids,
                                                    const float* __restrict__ tok,
                                                    float* __restrict__ x) {
  size_t idx = (size_t)blockIdx.x * 256 + threadIdx.x;
  int n = (int)(idx / Dd), d = (int)(idx % Dd);
  x[idx] = tok[(size_t)ids[n] * Dd + d];
}

// =====================================================================
// RMSNorm: out(f16) = x * rsqrt(mean(x^2)+eps) * w     one block per row
// =====================================================================
__global__ __launch_bounds__(256) void rmsnorm_kernel(const float* __restrict__ x,
                                                      const float* __restrict__ w,
                                                      _Float16* __restrict__ out) {
  int n = blockIdx.x, tid = threadIdx.x;
  const float* row = x + (size_t)n * Dd;
  float ss = 0.f;
#pragma unroll
  for (int it = 0; it < Dd / 256; ++it) { float v = row[tid + it * 256]; ss += v * v; }
#pragma unroll
  for (int off = 16; off > 0; off >>= 1) ss += __shfl_xor(ss, off, 32);
  __shared__ float sred[8];
  if ((tid & 31) == 0) sred[tid >> 5] = ss;
  __syncthreads();
  float tot = 0.f;
#pragma unroll
  for (int i = 0; i < 8; ++i) tot += sred[i];
  float rstd = rsqrtf(tot / (float)Dd + EPSf);
#pragma unroll
  for (int it = 0; it < Dd / 256; ++it) {
    int d = tid + it * 256;
    out[(size_t)n * Dd + d] = (_Float16)(row[d] * rstd * w[d]);
  }
}

// =====================================================================
// RoPE in-place on f16 [NTOK, H, HD] (rotate-half)
// =====================================================================
__global__ __launch_bounds__(256) void rope_kernel(_Float16* __restrict__ qk) {
  int idx = blockIdx.x * 256 + threadIdx.x;   // NTOK*H*32 threads
  int i  = idx & 31;
  int hh = (idx >> 5) & (Hh - 1);
  int n  = idx >> 9;
  int t  = n & (Tt - 1);
  float freq = __powf(10000.0f, -(float)i / 32.0f);
  float ang = (float)t * freq;
  float c = cosf(ang), s = sinf(ang);
  size_t base = (size_t)n * Dd + hh * HDd;
  float x1 = (float)qk[base + i], x2 = (float)qk[base + 32 + i];
  qk[base + i]      = (_Float16)(x1 * c - x2 * s);
  qk[base + 32 + i] = (_Float16)(x1 * s + x2 * c);
}

// =====================================================================
// Generic WMMA GEMM:  C[M,N] = A(f16)[M,K] x B(f32)[K,N] (or B[N,K] if bT)
//   - batched over z (aB/bB/cB element strides), optional per-z row count
//   - Cf(f32, optional accumulate) or Ch(f16)
//   BM=128 BN=128 BK=32, 256 threads (8 waves, each 64x32)
//   Software-pipelined: async A->LDS + register-staged B, double buffers.
// =====================================================================
__global__ __launch_bounds__(256) void gemm_kernel(
    const _Float16* __restrict__ A, long long aB,
    const float* __restrict__ Bm, long long bB,
    float* Cf, _Float16* Ch, long long cB,
    const int* __restrict__ cnt,
    int M, int N, int K, int lda, int ldb, int ldc, int bT, int accum) {
  const int z = blockIdx.z;
  A  += (size_t)z * aB;
  Bm += (size_t)z * bB;
  if (Cf) Cf += (size_t)z * cB;
  if (Ch) Ch += (size_t)z * cB;
  const int mrows = cnt ? cnt[z] : M;
  const int m0 = blockIdx.y * 128, n0 = blockIdx.x * 128;
  if (m0 >= mrows) return;

  __shared__ __align__(16) _Float16 As[2][128 * 40];
  __shared__ __align__(16) _Float16 Bs[2][128 * 40];   // stored [n][k]

  const int tid = threadIdx.x;
  const int w = tid >> 5, lane = tid & 31, half = lane >> 4, lr = lane & 15;
  const int wm = (w >> 2) * 64, wn = (w & 3) * 32;

  v8f acc[4][2];
#pragma unroll
  for (int i = 0; i < 4; ++i)
#pragma unroll
    for (int j = 0; j < 2; ++j) acc[i][j] = {};

  float4 breg[4];

  auto issueA = [&](int buf, int kb) {
#pragma unroll
    for (int it = 0; it < 2; ++it) {
      int idx = tid + it * 256;                 // 512 chunks of 8 halfs
      int r = idx >> 2, cq = idx & 3;
      int rg = m0 + r; if (rg >= mrows) rg = mrows - 1;
      async_copy_b128(lds_off_of(&As[buf][r * 40 + cq * 8]),
                      (const void*)(A + (size_t)rg * lda + kb + cq * 8));
    }
  };
  auto loadB = [&](int kb) {
    if (!bT) {
#pragma unroll
      for (int it = 0; it < 4; ++it) {
        int idx = tid + it * 256;               // 1024 float4
        int kk = idx >> 5, nq = idx & 31;
        const float* src = Bm + (size_t)(kb + kk) * ldb + n0 + nq * 4;
        breg[it] = *(const float4*)src;
        if (kb + 32 < K) __builtin_prefetch((const void*)(src + (size_t)32 * ldb), 0, 1);
      }
    } else {
#pragma unroll
      for (int it = 0; it < 4; ++it) {
        int idx = tid + it * 256;
        int nn = idx >> 3, kq = idx & 7;
        const float* src = Bm + (size_t)(n0 + nn) * ldb + kb + kq * 4;
        breg[it] = *(const float4*)src;
        if (kb + 32 < K) __builtin_prefetch((const void*)(src + 32), 0, 1);
      }
    }
  };
  auto storeB = [&](int buf) {
    if (!bT) {
#pragma unroll
      for (int it = 0; it < 4; ++it) {
        int idx = tid + it * 256;
        int kk = idx >> 5, nq = idx & 31;
        Bs[buf][(nq * 4 + 0) * 40 + kk] = (_Float16)breg[it].x;
        Bs[buf][(nq * 4 + 1) * 40 + kk] = (_Float16)breg[it].y;
        Bs[buf][(nq * 4 + 2) * 40 + kk] = (_Float16)breg[it].z;
        Bs[buf][(nq * 4 + 3) * 40 + kk] = (_Float16)breg[it].w;
      }
    } else {
#pragma unroll
      for (int it = 0; it < 4; ++it) {
        int idx = tid + it * 256;
        int nn = idx >> 3, kq = idx & 7;
        h4 hv; hv[0] = (_Float16)breg[it].x; hv[1] = (_Float16)breg[it].y;
        hv[2] = (_Float16)breg[it].z; hv[3] = (_Float16)breg[it].w;
        *(h4*)&Bs[buf][nn * 40 + kq * 4] = hv;
      }
    }
  };

  const int nk = K >> 5;
  issueA(0, 0);
  loadB(0);
  for (int kbi = 0; kbi < nk; ++kbi) {
    const int cur = kbi & 1, kb = kbi << 5;
    storeB(cur);
    wait_async0();
    __syncthreads();
    if (kbi + 1 < nk) { issueA(cur ^ 1, kb + 32); loadB(kb + 32); }
    // ---- fragments + 8 WMMAs per wave ----
    Frag af[4], bf[2];
#pragma unroll
    for (int i = 0; i < 4; ++i) {
      const h8* p = (const h8*)&As[cur][(wm + i * 16 + lr) * 40 + half * 8];
      af[i].h[0] = p[0]; af[i].h[1] = p[2];
    }
#pragma unroll
    for (int j = 0; j < 2; ++j) {
      const h8* p = (const h8*)&Bs[cur][(wn + j * 16 + lr) * 40 + half * 8];
      bf[j].h[0] = p[0]; bf[j].h[1] = p[2];
    }
#pragma unroll
    for (int i = 0; i < 4; ++i)
#pragma unroll
      for (int j = 0; j < 2; ++j) acc[i][j] = wmma16(af[i], bf[j], acc[i][j]);
    __syncthreads();
  }
  // ---- epilogue (C layout: vgpr r -> row half*8+r, col lr) ----
#pragma unroll
  for (int i = 0; i < 4; ++i)
#pragma unroll
    for (int j = 0; j < 2; ++j) {
      int rbase = m0 + wm + i * 16 + half * 8;
      int cc = n0 + wn + j * 16 + lr;
#pragma unroll
      for (int r = 0; r < 8; ++r) {
        int rr = rbase + r;
        if (rr < mrows) {
          float val = acc[i][j][r];
          if (Ch)          Ch[(size_t)rr * ldc + cc] = (_Float16)val;
          else if (accum)  Cf[(size_t)rr * ldc + cc] += val;
          else             Cf[(size_t)rr * ldc + cc] = val;
        }
      }
    }
}

// =====================================================================
// Flash-style attention with WMMA.
// grid (T/64, H, B), 128 threads (4 waves, each a 16-row q strip)
// q,k,v,o: f16 [NTOK, D] viewed as [b,t,h,hd]
// Pipelined: async K->LDS, register-staged transposed V, double buffers.
// =====================================================================
__global__ __launch_bounds__(128) void attn_kernel(const _Float16* __restrict__ q,
                                                   const _Float16* __restrict__ k,
                                                   const _Float16* __restrict__ v,
                                                   _Float16* __restrict__ o) {
  __shared__ __align__(16) _Float16 Ks[2][64 * 72];     // [t][hd]  (B^T layout)
  __shared__ __align__(16) _Float16 Vs[2][64 * 72];     // [hd][t]  (B^T layout)
  __shared__ __align__(16) _Float16 Ps[4 * 16 * 72];    // per-wave P strips

  const int tid = threadIdx.x;
  const int w = tid >> 5, lane = tid & 31, half = lane >> 4, lr = lane & 15;
  const int q0 = blockIdx.x * 64, hh = blockIdx.y, b = blockIdx.z;

  // Q fragments (A layout): row = w*16 + lr
  Frag qf[2];
  {
    size_t base = ((size_t)(b * Tt + q0 + w * 16 + lr)) * Dd + hh * HDd;
#pragma unroll
    for (int c = 0; c < 2; ++c) {
      const h8* p = (const h8*)(q + base + c * 32 + half * 8);
      qf[c].h[0] = p[0]; qf[c].h[1] = p[2];
    }
  }
  float m_i[8], l_i[8];
  v8f accO[4];
#pragma unroll
  for (int r = 0; r < 8; ++r) { m_i[r] = -3e38f; l_i[r] = 0.f; }
#pragma unroll
  for (int jn = 0; jn < 4; ++jn) accO[jn] = {};

  h8 vreg[4];
  auto issueK = [&](int buf, int k0) {
#pragma unroll
    for (int it = 0; it < 4; ++it) {
      int idx = tid + it * 128;                 // 512 chunks of 8 halfs
      int t = idx >> 3, co = idx & 7;
      size_t g = ((size_t)(b * Tt + k0 + t)) * Dd + hh * HDd + co * 8;
      async_copy_b128(lds_off_of(&Ks[buf][t * 72 + co * 8]), (const void*)(k + g));
    }
  };
  auto loadV = [&](int k0) {
#pragma unroll
    for (int it = 0; it < 4; ++it) {
      int idx = tid + it * 128;
      int t = idx >> 3, co = idx & 7;
      size_t g = ((size_t)(b * Tt + k0 + t)) * Dd + hh * HDd + co * 8;
      vreg[it] = *(const h8*)(v + g);
    }
  };
  auto storeV = [&](int buf) {
#pragma unroll
    for (int it = 0; it < 4; ++it) {
      int idx = tid + it * 128;
      int t = idx >> 3, co = idx & 7;
#pragma unroll
      for (int e = 0; e < 8; ++e) Vs[buf][(co * 8 + e) * 72 + t] = vreg[it][e];
    }
  };

  const int nkt = blockIdx.x + 1;              // causal: k-tiles up to q-tile
  issueK(0, 0);
  loadV(0);
  for (int kt = 0; kt < nkt; ++kt) {
    const int cur = kt & 1, k0 = kt * 64;
    storeV(cur);
    wait_async0();
    __syncthreads();
    if (kt + 1 < nkt) { issueK(cur ^ 1, k0 + 64); loadV(k0 + 64); }
    // ---- S = Q K^T ----
    v8f s[4];
#pragma unroll
    for (int jt = 0; jt < 4; ++jt) {
      s[jt] = {};
#pragma unroll
      for (int c = 0; c < 2; ++c) {
        Frag kf;
        const h8* p = (const h8*)&Ks[cur][(jt * 16 + lr) * 72 + c * 32 + half * 8];
        kf.h[0] = p[0]; kf.h[1] = p[2];
        s[jt] = wmma16(qf[c], kf, s[jt]);
      }
    }
    // ---- online softmax (half-wave reductions) ----
    const bool diag = (k0 == q0);
    float pv[4][8], mx[8];
#pragma unroll
    for (int r = 0; r < 8; ++r) {
      int qt = q0 + w * 16 + half * 8 + r;
      float m = -3e38f;
#pragma unroll
      for (int jt = 0; jt < 4; ++jt) {
        float val = s[jt][r] * 0.125f;
        if (diag && (k0 + jt * 16 + lr) > qt) val = -3e38f;
        pv[jt][r] = val;
        m = fmaxf(m, val);
      }
      mx[r] = m;
    }
#pragma unroll
    for (int r = 0; r < 8; ++r)
#pragma unroll
      for (int off = 1; off < 16; off <<= 1) mx[r] = fmaxf(mx[r], __shfl_xor(mx[r], off, 32));
    float sum[8];
#pragma unroll
    for (int r = 0; r < 8; ++r) {
      float mnew = fmaxf(m_i[r], mx[r]);
      float scl = __expf(m_i[r] - mnew);
      m_i[r] = mnew;
      float srow = 0.f;
#pragma unroll
      for (int jt = 0; jt < 4; ++jt) { float p = __expf(pv[jt][r] - mnew); pv[jt][r] = p; srow += p; }
      sum[r] = srow;
      l_i[r] *= scl;
#pragma unroll
      for (int jn = 0; jn < 4; ++jn) accO[jn][r] *= scl;
    }
#pragma unroll
    for (int r = 0; r < 8; ++r) {
#pragma unroll
      for (int off = 1; off < 16; off <<= 1) sum[r] += __shfl_xor(sum[r], off, 32);
      l_i[r] += sum[r];
    }
    // ---- P -> per-wave LDS strip (re-enter A-frag layout) ----
#pragma unroll
    for (int jt = 0; jt < 4; ++jt)
#pragma unroll
      for (int r = 0; r < 8; ++r)
        Ps[(w * 16 + half * 8 + r) * 72 + jt * 16 + lr] = (_Float16)pv[jt][r];
    // same-wave LDS ops are in-order: safe to read back now
    Frag pf[2];
#pragma unroll
    for (int c = 0; c < 2; ++c) {
      const h8* p = (const h8*)&Ps[(w * 16 + lr) * 72 + c * 32 + half * 8];
      pf[c].h[0] = p[0]; pf[c].h[1] = p[2];
    }
    // ---- O += P V ----
#pragma unroll
    for (int jn = 0; jn < 4; ++jn)
#pragma unroll
      for (int c = 0; c < 2; ++c) {
        Frag vf;
        const h8* p = (const h8*)&Vs[cur][(jn * 16 + lr) * 72 + c * 32 + half * 8];
        vf.h[0] = p[0]; vf.h[1] = p[2];
        accO[jn] = wmma16(pf[c], vf, accO[jn]);
      }
    __syncthreads();
  }
  // ---- write O (normalize by l) ----
#pragma unroll
  for (int jn = 0; jn < 4; ++jn)
#pragma unroll
    for (int r = 0; r < 8; ++r) {
      float ov = accO[jn][r] / l_i[r];
      size_t row = (size_t)(b * Tt + q0 + w * 16 + half * 8 + r);
      o[row * Dd + hh * HDd + jn * 16 + lr] = (_Float16)ov;
    }
}

// =====================================================================
// Router logits: logits[n,e] = xn[n,:] . router_w[:,e]   block per token
// =====================================================================
__global__ __launch_bounds__(256) void router_kernel(const _Float16* __restrict__ xn,
                                                     const float* __restrict__ rw,
                                                     float* __restrict__ logits) {
  __shared__ float sm[256 * Ee];
  int n = blockIdx.x, tid = threadIdx.x;
  float acc[Ee];
#pragma unroll
  for (int e = 0; e < Ee; ++e) acc[e] = 0.f;
  for (int d = tid; d < Dd; d += 256) {
    float xv = (float)xn[(size_t)n * Dd + d];
    const float* r = rw + (size_t)d * Ee;
#pragma unroll
    for (int e = 0; e < Ee; ++e) acc[e] += xv * r[e];
  }
#pragma unroll
  for (int e = 0; e < Ee; ++e) sm[tid * Ee + e] = acc[e];
  __syncthreads();
  for (int s = 128; s > 0; s >>= 1) {
    if (tid < s)
#pragma unroll
      for (int e = 0; e < Ee; ++e) sm[tid * Ee + e] += sm[(tid + s) * Ee + e];
    __syncthreads();
  }
  if (tid < Ee) logits[(size_t)n * Ee + tid] = sm[tid];
}

// =====================================================================
// Top-2 + softmax gates
// =====================================================================
__global__ __launch_bounds__(256) void topk_kernel(const float* __restrict__ logits,
                                                   int2* __restrict__ tope,
                                                   float2* __restrict__ topw) {
  int n = blockIdx.x * 256 + threadIdx.x;
  float lv[Ee];
#pragma unroll
  for (int e = 0; e < Ee; ++e) lv[e] = logits[(size_t)n * Ee + e];
  int i0 = 0;
#pragma unroll
  for (int e = 1; e < Ee; ++e) if (lv[e] > lv[i0]) i0 = e;
  int i1 = (i0 == 0) ? 1 : 0;
#pragma unroll
  for (int e = 0; e < Ee; ++e) if (e != i0 && lv[e] > lv[i1]) i1 = e;
  float w0 = 1.f / (1.f + __expf(lv[i1] - lv[i0]));
  int2 te; te.x = i0; te.y = i1;
  float2 tw; tw.x = w0; tw.y = 1.f - w0;
  tope[n] = te; topw[n] = tw;
}

// =====================================================================
// Deterministic per-expert token lists (serial scan per expert)
// =====================================================================
__global__ void build_lists_kernel(const int2* __restrict__ tope,
                                   int* __restrict__ perm,
                                   int* __restrict__ slot,
                                   int* __restrict__ counts) {
  int e = threadIdx.x;
  if (e >= Ee) return;
  int c = 0;
  for (int n = 0; n < NTOK; ++n) {
    int2 t = tope[n];
    if (t.x == e)      { perm[e * NTOK + c] = n; slot[2 * n + 0] = c; ++c; }
    else if (t.y == e) { perm[e * NTOK + c] = n; slot[2 * n + 1] = c; ++c; }
  }
  counts[e] = c;
}

// =====================================================================
// MoE FFN stage 1 (fused, gathered): h = silu(x Wg) * (x Wu)   f16 out
// grid (HID/64, NTOK/128, E), 256 threads; BM=128 BN=64 BK=32
// Pipelined: async gathered A->LDS + register-staged Wg/Wu, double buffers.
// =====================================================================
__global__ __launch_bounds__(256) void ffn1_kernel(const _Float16* __restrict__ xn,
                                                   const float* __restrict__ Wg,
                                                   const float* __restrict__ Wu,
                                                   const int* __restrict__ perm,
                                                   const int* __restrict__ counts,
                                                   _Float16* __restrict__ hbuf) {
  const int z = blockIdx.z;
  const int cnt = counts[z];
  const int m0 = blockIdx.y * 128, n0 = blockIdx.x * 64;
  if (m0 >= cnt) return;
  Wg += (size_t)z * Dd * HIDd;
  Wu += (size_t)z * Dd * HIDd;
  const int* gp = perm + z * NTOK;

  __shared__ __align__(16) _Float16 As[2][128 * 40];
  __shared__ __align__(16) _Float16 Bg[2][64 * 40];
  __shared__ __align__(16) _Float16 Bu[2][64 * 40];

  const int tid = threadIdx.x;
  const int w = tid >> 5, lane = tid & 31, half = lane >> 4, lr = lane & 15;
  const int wm = (w >> 1) * 32, wn = (w & 1) * 32;

  v8f accG[2][2], accU[2][2];
#pragma unroll
  for (int i = 0; i < 2; ++i)
#pragma unroll
    for (int j = 0; j < 2; ++j) { accG[i][j] = {}; accU[i][j] = {}; }

  float4 greg[2], ureg[2];
  auto issueA = [&](int buf, int kb) {
#pragma unroll
    for (int it = 0; it < 2; ++it) {
      int idx = tid + it * 256;
      int r = idx >> 2, cq = idx & 3;
      int rg = m0 + r; if (rg >= cnt) rg = cnt - 1;
      int arow = gp[rg];
      async_copy_b128(lds_off_of(&As[buf][r * 40 + cq * 8]),
                      (const void*)(xn + (size_t)arow * Dd + kb + cq * 8));
    }
  };
  auto loadBgu = [&](int kb) {
#pragma unroll
    for (int it = 0; it < 2; ++it) {
      int idx = tid + it * 256;                 // 512 float4 per matrix
      int kk = idx >> 4, nq = idx & 15;
      const float* sg = Wg + (size_t)(kb + kk) * HIDd + n0 + nq * 4;
      const float* su = Wu + (size_t)(kb + kk) * HIDd + n0 + nq * 4;
      greg[it] = *(const float4*)sg;
      ureg[it] = *(const float4*)su;
      if (kb + 32 < Dd) {
        __builtin_prefetch((const void*)(sg + (size_t)32 * HIDd), 0, 1);
        __builtin_prefetch((const void*)(su + (size_t)32 * HIDd), 0, 1);
      }
    }
  };
  auto storeBgu = [&](int buf) {
#pragma unroll
    for (int it = 0; it < 2; ++it) {
      int idx = tid + it * 256;
      int kk = idx >> 4, nq = idx & 15;
      Bg[buf][(nq * 4 + 0) * 40 + kk] = (_Float16)greg[it].x;
      Bg[buf][(nq * 4 + 1) * 40 + kk] = (_Float16)greg[it].y;
      Bg[buf][(nq * 4 + 2) * 40 + kk] = (_Float16)greg[it].z;
      Bg[buf][(nq * 4 + 3) * 40 + kk] = (_Float16)greg[it].w;
      Bu[buf][(nq * 4 + 0) * 40 + kk] = (_Float16)ureg[it].x;
      Bu[buf][(nq * 4 + 1) * 40 + kk] = (_Float16)ureg[it].y;
      Bu[buf][(nq * 4 + 2) * 40 + kk] = (_Float16)ureg[it].z;
      Bu[buf][(nq * 4 + 3) * 40 + kk] = (_Float16)ureg[it].w;
    }
  };

  const int nk = Dd / 32;
  issueA(0, 0);
  loadBgu(0);
  for (int kbi = 0; kbi < nk; ++kbi) {
    const int cur = kbi & 1, kb = kbi << 5;
    storeBgu(cur);
    wait_async0();
    __syncthreads();
    if (kbi + 1 < nk) { issueA(cur ^ 1, kb + 32); loadBgu(kb + 32); }
    Frag af[2], bgf[2], buf2[2];
#pragma unroll
    for (int i = 0; i < 2; ++i) {
      const h8* p = (const h8*)&As[cur][(wm + i * 16 + lr) * 40 + half * 8];
      af[i].h[0] = p[0]; af[i].h[1] = p[2];
    }
#pragma unroll
    for (int j = 0; j < 2; ++j) {
      const h8* pg = (const h8*)&Bg[cur][(wn + j * 16 + lr) * 40 + half * 8];
      bgf[j].h[0] = pg[0]; bgf[j].h[1] = pg[2];
      const h8* pu = (const h8*)&Bu[cur][(wn + j * 16 + lr) * 40 + half * 8];
      buf2[j].h[0] = pu[0]; buf2[j].h[1] = pu[2];
    }
#pragma unroll
    for (int i = 0; i < 2; ++i)
#pragma unroll
      for (int j = 0; j < 2; ++j) {
        accG[i][j] = wmma16(af[i], bgf[j], accG[i][j]);
        accU[i][j] = wmma16(af[i], buf2[j], accU[i][j]);
      }
    __syncthreads();
  }
#pragma unroll
  for (int i = 0; i < 2; ++i)
#pragma unroll
    for (int j = 0; j < 2; ++j) {
      int rbase = m0 + wm + i * 16 + half * 8;
      int cc = n0 + wn + j * 16 + lr;
#pragma unroll
      for (int r = 0; r < 8; ++r) {
        int rr = rbase + r;
        if (rr < cnt) {
          float g = accG[i][j][r], u = accU[i][j][r];
          float hval = g / (1.f + __expf(-g)) * u;   // silu(g)*u
          hbuf[((size_t)z * NTOK + rr) * HIDd + cc] = (_Float16)hval;
        }
      }
    }
}

// =====================================================================
// MoE combine (deterministic): x[n] += w0*y[e0,s0] + w1*y[e1,s1]
// =====================================================================
__global__ __launch_bounds__(256) void combine_kernel(float* __restrict__ x,
                                                      const _Float16* __restrict__ y,
                                                      const int2* __restrict__ tope,
                                                      const float2* __restrict__ topw,
                                                      const int* __restrict__ slot) {
  size_t idx = (size_t)blockIdx.x * 256 + threadIdx.x;
  int n = (int)(idx / Dd), d = (int)(idx % Dd);
  int2 te = tope[n]; float2 tw = topw[n];
  int s0 = slot[2 * n + 0], s1 = slot[2 * n + 1];
  float v0 = (float)y[((size_t)te.x * NTOK + s0) * Dd + d];
  float v1 = (float)y[((size_t)te.y * NTOK + s1) * Dd + d];
  x[idx] += tw.x * v0 + tw.y * v1;
}

// =====================================================================
// Host launcher
// =====================================================================
extern "C" void kernel_launch(void* const* d_in, const int* in_sizes, int n_in,
                              void* d_out, int out_size, void* d_ws, size_t ws_size,
                              hipStream_t stream) {
  (void)in_sizes; (void)n_in; (void)out_size; (void)ws_size;
  const int*   ids = (const int*)d_in[0];
  const float* tok = (const float*)d_in[1];
  const float* anw = (const float*)d_in[2];
  const float* wq  = (const float*)d_in[3];
  const float* wk  = (const float*)d_in[4];
  const float* wv  = (const float*)d_in[5];
  const float* wo  = (const float*)d_in[6];
  const float* mnw = (const float*)d_in[7];
  const float* rw  = (const float*)d_in[8];
  const float* wg  = (const float*)d_in[9];
  const float* wu  = (const float*)d_in[10];
  const float* wd  = (const float*)d_in[11];
  const float* fnw = (const float*)d_in[12];

  char* p = (char*)d_ws;
  auto alloc = [&](size_t bytes) { char* r = p; p += (bytes + 255) & ~(size_t)255; return r; };
  float*     x      = (float*)    alloc((size_t)NTOK * Dd * 4);
  _Float16*  xn     = (_Float16*) alloc((size_t)NTOK * Dd * 2);
  _Float16*  qb     = (_Float16*) alloc((size_t)NTOK * Dd * 2);
  _Float16*  kb     = (_Float16*) alloc((size_t)NTOK * Dd * 2);
  _Float16*  vb     = (_Float16*) alloc((size_t)NTOK * Dd * 2);
  _Float16*  ob     = (_Float16*) alloc((size_t)NTOK * Dd * 2);
  float*     logits = (float*)    alloc((size_t)NTOK * Ee * 4);
  int2*      tope   = (int2*)     alloc((size_t)NTOK * 8);
  float2*    topw   = (float2*)   alloc((size_t)NTOK * 8);
  int*       perm   = (int*)      alloc((size_t)Ee * NTOK * 4);
  int*       slot   = (int*)      alloc((size_t)NTOK * 2 * 4);
  int*       counts = (int*)      alloc(256);
  _Float16*  hbuf   = (_Float16*) alloc((size_t)Ee * NTOK * HIDd * 2);
  _Float16*  ybuf   = (_Float16*) alloc((size_t)Ee * NTOK * Dd * 2);

  embed_kernel<<<(NTOK * Dd) / 256, 256, 0, stream>>>(ids, tok, x);

  const dim3 gNN(Dd / 128, NTOK / 128, 1);
  for (int l = 0; l < Ll; ++l) {
    // ---- attention block ----
    rmsnorm_kernel<<<NTOK, 256, 0, stream>>>(x, anw + (size_t)l * Dd, xn);
    gemm_kernel<<<gNN, 256, 0, stream>>>(xn, 0, wq + (size_t)l * Dd * Dd, 0,
        nullptr, qb, 0, nullptr, NTOK, Dd, Dd, Dd, Dd, Dd, 0, 0);
    gemm_kernel<<<gNN, 256, 0, stream>>>(xn, 0, wk + (size_t)l * Dd * Dd, 0,
        nullptr, kb, 0, nullptr, NTOK, Dd, Dd, Dd, Dd, Dd, 0, 0);
    gemm_kernel<<<gNN, 256, 0, stream>>>(xn, 0, wv + (size_t)l * Dd * Dd, 0,
        nullptr, vb, 0, nullptr, NTOK, Dd, Dd, Dd, Dd, Dd, 0, 0);
    rope_kernel<<<(NTOK * Hh * 32) / 256, 256, 0, stream>>>(qb);
    rope_kernel<<<(NTOK * Hh * 32) / 256, 256, 0, stream>>>(kb);
    attn_kernel<<<dim3(Tt / 64, Hh, Bb), 128, 0, stream>>>(qb, kb, vb, ob);
    gemm_kernel<<<gNN, 256, 0, stream>>>(ob, 0, wo + (size_t)l * Dd * Dd, 0,
        x, nullptr, 0, nullptr, NTOK, Dd, Dd, Dd, Dd, Dd, 0, 1);   // x += o @ wo

    // ---- MoE block ----
    rmsnorm_kernel<<<NTOK, 256, 0, stream>>>(x, mnw + (size_t)l * Dd, xn);
    router_kernel<<<NTOK, 256, 0, stream>>>(xn, rw + (size_t)l * Dd * Ee, logits);
    topk_kernel<<<NTOK / 256, 256, 0, stream>>>(logits, tope, topw);
    build_lists_kernel<<<1, 32, 0, stream>>>(tope, perm, slot, counts);
    ffn1_kernel<<<dim3(HIDd / 64, NTOK / 128, Ee), 256, 0, stream>>>(
        xn, wg + (size_t)l * Ee * Dd * HIDd, wu + (size_t)l * Ee * Dd * HIDd,
        perm, counts, hbuf);
    gemm_kernel<<<dim3(Dd / 128, NTOK / 128, Ee), 256, 0, stream>>>(
        hbuf, (long long)NTOK * HIDd,
        wd + (size_t)l * Ee * HIDd * Dd, (long long)HIDd * Dd,
        nullptr, ybuf, (long long)NTOK * Dd, counts,
        NTOK, Dd, HIDd, HIDd, Dd, Dd, 0, 0);
    combine_kernel<<<(NTOK * Dd) / 256, 256, 0, stream>>>(x, ybuf, tope, topw, slot);
  }

  // ---- final norm + tied LM head (B^T = tok_embed [V, D]) ----
  rmsnorm_kernel<<<NTOK, 256, 0, stream>>>(x, fnw, xn);
  gemm_kernel<<<dim3(Vv / 128, NTOK / 128, 1), 256, 0, stream>>>(
      xn, 0, tok, 0, (float*)d_out, nullptr, 0, nullptr,
      NTOK, Vv, Dd, Dd, Dd, Vv, 1, 0);
}
